// WikiSQLModel_53755810676911
// MI455X (gfx1250) — matrix operand order repaired and checked by reading, not compiled, not run on hardware
//
#include <hip/hip_runtime.h>
#include <hip/hip_bf16.h>

// ---------------------------------------------------------------------------
// CDNA5 (gfx1250) bf16 WMMA implementation of the WikiSQL cross-attention fwd.
// All GEMMs via v_wmma_f32_16x16x32_bf16 (wave32). f32 accumulation.
// Register-pipelined K-loops so fragment loads overlap WMMA issue.
// ---------------------------------------------------------------------------

typedef __attribute__((ext_vector_type(16))) __bf16 v16bf;
typedef __attribute__((ext_vector_type(8)))  __bf16 bf16x8;
typedef __attribute__((ext_vector_type(8)))  float  v8f;

static constexpr int Bsz = 64;
static constexpr int S   = 512;
static constexpr int H   = 768;
static constexpr int L   = 511;   // sliced length
static constexpr int NH  = 8;
static constexpr int HD  = 96;

__device__ __forceinline__ v16bf make_frag(bf16x8 lo, bf16x8 hi) {
    v16bf a;
#pragma unroll
    for (int e = 0; e < 8; ++e) { a[e] = lo[e]; a[e + 8] = hi[e]; }
    return a;
}

// A fragment (16x32 bf16) from a row-major [16][stride] tile.
// Per ISA 7.12.2: lane r (r=lane&15, half=lane>>4) holds row r,
// K = {k0+half*8 .. +7} and {k0+16+half*8 .. +7}  -> two b128 loads.
__device__ __forceinline__ v16bf load_a(const __bf16* base, int stride, int k0, int lane) {
    int r = lane & 15, h = lane >> 4;
    const __bf16* row = base + (size_t)r * stride + k0 + h * 8;
    bf16x8 lo = *(const bf16x8*)(row);
    bf16x8 hi = *(const bf16x8*)(row + 16);
    return make_frag(lo, hi);
}

// B fragment (32x16 bf16): column n = n0+(lane&15), per lane
// K = {k0+half*8 .. +7} and {k0+16+half*8 .. +7}  -> two b128 loads.
__device__ __forceinline__ v16bf load_b(const __bf16* base, int stride, int n0, int k0, int lane) {
    int n = n0 + (lane & 15), h = lane >> 4;
    const __bf16* p = base + (size_t)n * stride + k0 + h * 8;
    bf16x8 lo = *(const bf16x8*)(p);
    bf16x8 hi = *(const bf16x8*)(p + 16);
    return make_frag(lo, hi);
}

__device__ __forceinline__ v8f wmma_bf16(v16bf a, v16bf b, v8f c) {
    return __builtin_amdgcn_wmma_f32_16x16x32_bf16(false, a, false, b, (short)0, c, false, false);
}

// Register-pipelined 768-K GEMM tile: A row-major [16][768], B col n0..n0+15.
__device__ __forceinline__ v8f gemm_k768(const __bf16* A, const __bf16* W, int n0, int lane) {
    v8f acc = {};
    v16bf af = load_a(A, H, 0, lane);
    v16bf bf = load_b(W, H, n0, 0, lane);
#pragma unroll 4
    for (int k0 = 32; k0 < H; k0 += 32) {
        v16bf af2 = load_a(A, H, k0, lane);      // next-step loads in flight
        v16bf bf2 = load_b(W, H, n0, k0, lane);  // while WMMA consumes current
        acc = wmma_bf16(af, bf, acc);
        af = af2; bf = bf2;
    }
    return wmma_bf16(af, bf, acc);
}

// ---------------------------------------------------------------------------
// Kernel W: convert f32 weights to bf16 once (weights then live in L2).
// ---------------------------------------------------------------------------
__global__ void cvt_kernel(const float* __restrict__ wq, const float* __restrict__ wk,
                           const float* __restrict__ wv, const float* __restrict__ wo,
                           __bf16* __restrict__ Wqkv, __bf16* __restrict__ Wob) {
    int i = blockIdx.x * 256 + threadIdx.x;
    if (i >= H * H) return;
    Wqkv[i]             = (__bf16)wq[i];
    Wqkv[i + H * H]     = (__bf16)wk[i];
    Wqkv[i + 2 * H * H] = (__bf16)wv[i];
    Wob[i]              = (__bf16)wo[i];
}

__global__ void zero_kernel(float* p, int n) {
    int i = blockIdx.x * 256 + threadIdx.x;
    if (i < n) p[i] = 0.f;
}

// ---------------------------------------------------------------------------
// Kernel A: fused QKV projection.  grid=(32 row-tiles, 64 batches), 256 thr.
// Q,K stored row-major bf16 [B][512][768]; V stored transposed [B][768][512].
// ---------------------------------------------------------------------------
__global__ void __launch_bounds__(256) qkv_kernel(
    const float* __restrict__ text, const float* __restrict__ cols,
    const __bf16* __restrict__ Wqkv,
    const float* __restrict__ bq, const float* __restrict__ bk, const float* __restrict__ bv,
    __bf16* __restrict__ Qb, __bf16* __restrict__ Kb, __bf16* __restrict__ Vt) {
    __shared__ __align__(16) __bf16 Ac[16][H];  // cols tile (for Q)
    __shared__ __align__(16) __bf16 At[16][H];  // text tile (for K,V)
    const int tile = blockIdx.x, bb = blockIdx.y, tid = threadIdx.x;

    for (int idx = tid; idx < 16 * H; idx += 256) {
        int r = idx / H, k = idx - r * H;
        int l = tile * 16 + r;
        float fc = 0.f, ft = 0.f;
        if (l < L) {                       // padded row 511 -> zeros
            size_t o = ((size_t)bb * S + (l + 1)) * H + k;   // slice offset +1
            fc = cols[o]; ft = text[o];
        }
        Ac[r][k] = (__bf16)fc;
        At[r][k] = (__bf16)ft;
    }
    __syncthreads();

    const int lane = tid & 31, wave = tid >> 5;
    for (int t = wave; t < 3 * 48; t += 8) {
        const int m = t / 48;          // 0=Q 1=K 2=V
        const int j = t - m * 48;      // output column tile
        const __bf16* A = (m == 0) ? &Ac[0][0] : &At[0][0];
        const __bf16* W = Wqkv + (size_t)m * H * H;
        v8f acc = gemm_k768(A, W, j * 16, lane);

        const int n = j * 16 + (lane & 15);
        const int h = lane >> 4;
        const float* bias = (m == 0) ? bq : ((m == 1) ? bk : bv);
        const float bval = bias[n];
        if (m == 2) {
            __bf16* O = Vt + ((size_t)bb * H + n) * S + tile * 16 + h * 8;
#pragma unroll
            for (int r2 = 0; r2 < 8; ++r2) O[r2] = (__bf16)(acc[r2] + bval);      // transposed
        } else {
            __bf16* O = ((m == 0) ? Qb : Kb) + ((size_t)bb * S + tile * 16 + h * 8) * H + n;
#pragma unroll
            for (int r2 = 0; r2 < 8; ++r2) O[(size_t)r2 * H] = (__bf16)(acc[r2] + bval);
        }
    }
}

// ---------------------------------------------------------------------------
// Kernel B: attention per (query-tile, head, batch). One wave per block.
// scores (16x512) -> LDS -> masked softmax -> bf16 probs -> ctx (16x96).
// ---------------------------------------------------------------------------
__global__ void __launch_bounds__(32) attn_kernel(
    const __bf16* __restrict__ Qb, const __bf16* __restrict__ Kb,
    const __bf16* __restrict__ Vt, __bf16* __restrict__ Ctx) {
    __shared__ __align__(16) float  sc[16][S];
    __shared__ __align__(16) __bf16 pr[16][S];
    __shared__ float rowsum[16];
    const int qtile = blockIdx.x, hh = blockIdx.y, bb = blockIdx.z;
    const int lane = threadIdx.x;
    const int r = lane & 15, h = lane >> 4;
    const float scale = 0.1020620726159658f;  // 1/sqrt(96)

    // Q A-fragments (16x96 = 3 K-windows), reused over all 32 kpos tiles.
    const __bf16* Qrow = Qb + ((size_t)bb * S + qtile * 16) * H + hh * HD;
    v16bf aq0 = load_a(Qrow, H, 0,  lane);
    v16bf aq1 = load_a(Qrow, H, 32, lane);
    v16bf aq2 = load_a(Qrow, H, 64, lane);

    // scores: pipeline next kpos tile's K fragments under current 3 WMMAs.
    const __bf16* Kb0 = Kb + ((size_t)bb * S) * H + hh * HD;
    v16bf b0 = load_b(Kb0, H, 0, 0,  lane);
    v16bf b1 = load_b(Kb0, H, 0, 32, lane);
    v16bf b2 = load_b(Kb0, H, 0, 64, lane);
    for (int kt = 0; kt < 32; ++kt) {
        const int ktn = (kt + 1 < 32) ? kt + 1 : kt;      // uniform clamp (no divergence)
        const __bf16* Kn = Kb0 + (size_t)ktn * 16 * H;
        v16bf n0 = load_b(Kn, H, 0, 0,  lane);
        v16bf n1 = load_b(Kn, H, 0, 32, lane);
        v16bf n2 = load_b(Kn, H, 0, 64, lane);
        v8f acc = {};
        acc = wmma_bf16(aq0, b0, acc);
        acc = wmma_bf16(aq1, b1, acc);
        acc = wmma_bf16(aq2, b2, acc);
        const int kp = kt * 16 + r;
#pragma unroll
        for (int r2 = 0; r2 < 8; ++r2) {
            int m = r2 + h * 8;
            sc[m][kp] = (kp < L) ? acc[r2] * scale : -1e30f;  // mask pad kpos
        }
        b0 = n0; b1 = n1; b2 = n2;
    }
    __syncthreads();

    // softmax over 512: lane pair (r, r+16) split columns, combine via shfl_xor.
    const int c0 = h * 256;
    float mx = -3.4e38f;
    for (int c = 0; c < 256; ++c) mx = fmaxf(mx, sc[r][c0 + c]);
    mx = fmaxf(mx, __shfl_xor(mx, 16));
    float sum = 0.f;
    for (int c = 0; c < 256; ++c) {
        float e = __expf(sc[r][c0 + c] - mx);
        pr[r][c0 + c] = (__bf16)e;     // unnormalized; fold 1/sum into ctx
        sum += e;
    }
    sum += __shfl_xor(sum, 16);
    if (h == 0) rowsum[r] = sum;
    __syncthreads();

    // ctx = P @ V  (K = 512 kpos, contiguous along Vt rows), pipelined.
    for (int dt = 0; dt < 6; ++dt) {
        const __bf16* Vbase = Vt + ((size_t)bb * H + hh * HD + dt * 16) * S;
        v8f acc = {};
        v16bf ap = load_a(&pr[0][0], S, 0, lane);
        v16bf bv = load_b(Vbase, S, 0, 0, lane);
#pragma unroll 4
        for (int k0 = 32; k0 < S; k0 += 32) {
            v16bf ap2 = load_a(&pr[0][0], S, k0, lane);
            v16bf bv2 = load_b(Vbase, S, 0, k0, lane);
            acc = wmma_bf16(ap, bv, acc);
            ap = ap2; bv = bv2;
        }
        acc = wmma_bf16(ap, bv, acc);

        const int n = hh * HD + dt * 16 + r;
#pragma unroll
        for (int r2 = 0; r2 < 8; ++r2) {
            int m = r2 + h * 8;
            float val = acc[r2] * (1.f / rowsum[m]);
            Ctx[((size_t)bb * S + qtile * 16 + m) * H + n] = (__bf16)val;
        }
    }
}

// ---------------------------------------------------------------------------
// Kernel C: output projection + per-channel (B,L) sum/sumsq for the layernorm.
// ---------------------------------------------------------------------------
__global__ void __launch_bounds__(256) oproj_kernel(
    const __bf16* __restrict__ Ctx, const __bf16* __restrict__ Wob,
    const float* __restrict__ bo, float* __restrict__ AO,
    float* __restrict__ gsum, float* __restrict__ gsq) {
    __shared__ __align__(16) __bf16 Atile[16][H];
    __shared__ float ssum[H], ssq[H];
    const int tile = blockIdx.x, bb = blockIdx.y, tid = threadIdx.x;

    for (int i = tid; i < H; i += 256) { ssum[i] = 0.f; ssq[i] = 0.f; }
    for (int idx = tid; idx < 16 * H; idx += 256) {
        int r = idx / H, k = idx - r * H;
        Atile[r][k] = Ctx[((size_t)bb * S + tile * 16 + r) * H + k];
    }
    __syncthreads();

    const int lane = tid & 31, wave = tid >> 5;
    for (int j = wave; j < 48; j += 8) {
        v8f acc = gemm_k768(&Atile[0][0], Wob, j * 16, lane);
        const int n = j * 16 + (lane & 15);
        const int h = lane >> 4;
        const float bias = bo[n];
#pragma unroll
        for (int r2 = 0; r2 < 8; ++r2) {
            int l = tile * 16 + r2 + h * 8;
            float val = acc[r2] + bias;
            AO[((size_t)bb * S + l) * H + n] = val;
            if (l < L) {                      // exclude pad row from stats
                atomicAdd(&ssum[n], val);
                atomicAdd(&ssq[n], val * val);
            }
        }
    }
    __syncthreads();
    for (int i = tid; i < H; i += 256) {
        atomicAdd(&gsum[i], ssum[i]);
        atomicAdd(&gsq[i], ssq[i]);
    }
}

// ---------------------------------------------------------------------------
// Kernel D: fold mu/rstd/gamma/beta/w_out into per-channel coefficients:
//   fv = base + sum_ch ( w_out[ch]*c[ch] + aCoef[ch]*AO[ch] )
// ---------------------------------------------------------------------------
__global__ void stats_kernel(const float* __restrict__ gsum, const float* __restrict__ gsq,
                             const float* __restrict__ gamma, const float* __restrict__ beta,
                             const float* __restrict__ w_out, const float* __restrict__ b_out,
                             float* __restrict__ aCoef, float* __restrict__ basep) {
    __shared__ float red[256];
    const int tid = threadIdx.x;
    const float N = (float)Bsz * (float)L;
    float kacc = 0.f;
    for (int ch = tid; ch < H; ch += 256) {
        float mu   = gsum[ch] / N;
        float var  = gsq[ch] / N - mu * mu;
        float rstd = rsqrtf(var + 1e-5f);
        float g = gamma[ch], w = w_out[ch];
        aCoef[ch] = w * rstd * g;
        kacc += w * (beta[ch] - mu * rstd * g);
    }
    red[tid] = kacc;
    __syncthreads();
    for (int s = 128; s > 0; s >>= 1) { if (tid < s) red[tid] += red[tid + s]; __syncthreads(); }
    if (tid == 0) basep[0] = red[0] + b_out[0];
}

// ---------------------------------------------------------------------------
// Kernel E: fv[b][l] — one warp per row, 768-dot with warp shuffle reduction.
// ---------------------------------------------------------------------------
__global__ void __launch_bounds__(256) fv_kernel(
    const float* __restrict__ cols, const float* __restrict__ AO,
    const float* __restrict__ w_out, const float* __restrict__ aCoef,
    const float* __restrict__ basep, float* __restrict__ fv) {
    const int wave = threadIdx.x >> 5, lane = threadIdx.x & 31;
    const int row = blockIdx.x * 8 + wave;
    const int bb = row >> 9, l = row & (S - 1);
    if (l >= L) return;
    const float* cr = cols + ((size_t)bb * S + l + 1) * H;
    const float* ar = AO + ((size_t)bb * S + l) * H;
    float acc = 0.f;
    for (int ch = lane; ch < H; ch += 32)
        acc += w_out[ch] * cr[ch] + aCoef[ch] * ar[ch];
    for (int o = 16; o > 0; o >>= 1) acc += __shfl_down(acc, o);
    if (lane == 0) fv[row] = basep[0] + acc;
}

// ---------------------------------------------------------------------------
// Kernel F: comma-segment mean pooling, one thread per batch (serial cumsum).
// ---------------------------------------------------------------------------
__global__ void pool_kernel(const float* __restrict__ fv, const int* __restrict__ ids,
                            float* __restrict__ out) {
    const int bb = threadIdx.x;
    if (bb >= Bsz) return;
    float sums[17], cnt[17];
    for (int j = 0; j < 17; ++j) { sums[j] = 0.f; cnt[j] = 0.f; }
    int commas = 0;
    for (int l = 0; l < L; ++l) {
        int isc = (ids[bb * S + l] == 1010) ? 1 : 0;
        if (l >= 1 && !isc && commas < 17) {
            sums[commas] += fv[bb * S + l];
            cnt[commas] += 1.f;
        }
        commas += isc;
    }
    for (int j = 0; j < 17; ++j) out[bb * 17 + j] = sums[j] / fmaxf(cnt[j], 1.f);
}

// ---------------------------------------------------------------------------
extern "C" void kernel_launch(void* const* d_in, const int* in_sizes, int n_in,
                              void* d_out, int out_size, void* d_ws, size_t ws_size,
                              hipStream_t stream) {
    const float* text  = (const float*)d_in[0];
    const float* cols  = (const float*)d_in[1];
    const float* wq    = (const float*)d_in[2];
    const float* wk    = (const float*)d_in[3];
    const float* wv    = (const float*)d_in[4];
    const float* bq    = (const float*)d_in[5];
    const float* bk    = (const float*)d_in[6];
    const float* bv    = (const float*)d_in[7];
    const float* wo    = (const float*)d_in[8];
    const float* bo    = (const float*)d_in[9];
    const float* gamma = (const float*)d_in[10];
    const float* beta  = (const float*)d_in[11];
    const float* w_out = (const float*)d_in[12];
    const float* b_out = (const float*)d_in[13];
    const int*   ids   = (const int*)d_in[14];
    float* out = (float*)d_out;

    // carve workspace (256B-aligned)
    size_t off = 0;
    auto alloc = [&](size_t bytes) {
        void* p = (char*)d_ws + off;
        off = (off + bytes + 255) & ~(size_t)255;
        return p;
    };
    const size_t actE = (size_t)Bsz * S * H;          // 25,165,824
    __bf16* Qb   = (__bf16*)alloc(actE * 2);
    __bf16* Kb   = (__bf16*)alloc(actE * 2);
    __bf16* Vt   = (__bf16*)alloc(actE * 2);
    __bf16* Ctx  = (__bf16*)alloc(actE * 2);
    float*  AO   = (float*)alloc(actE * 4);
    __bf16* Wqkv = (__bf16*)alloc((size_t)3 * H * H * 2);
    __bf16* Wob  = (__bf16*)alloc((size_t)H * H * 2);
    float*  gsum = (float*)alloc(H * 4);
    float*  gsq  = (float*)alloc(H * 4);
    float*  aCoef= (float*)alloc(H * 4);
    float*  basep= (float*)alloc(256);
    float*  fv   = (float*)alloc((size_t)Bsz * S * 4);
    (void)ws_size; (void)in_sizes; (void)n_in; (void)out_size;

    cvt_kernel<<<(H * H + 255) / 256, 256, 0, stream>>>(wq, wk, wv, wo, Wqkv, Wob);
    zero_kernel<<<(2 * H + 255) / 256, 256, 0, stream>>>(gsum, 2 * H);  // gsum+gsq contiguous
    qkv_kernel<<<dim3(32, Bsz), 256, 0, stream>>>(text, cols, Wqkv, bq, bk, bv, Qb, Kb, Vt);
    attn_kernel<<<dim3(32, NH, Bsz), 32, 0, stream>>>(Qb, Kb, Vt, Ctx);
    oproj_kernel<<<dim3(32, Bsz), 256, 0, stream>>>(Ctx, Wob, bo, AO, gsum, gsq);
    stats_kernel<<<1, 256, 0, stream>>>(gsum, gsq, gamma, beta, w_out, b_out, aCoef, basep);
    fv_kernel<<<(Bsz * S) / 8, 256, 0, stream>>>(cols, AO, w_out, aCoef, basep, fv);
    pool_kernel<<<1, 64, 0, stream>>>(fv, ids, out);
}
